// MultiHeadSelfAttention_25434796327134
// MI455X (gfx1250) — compile-verified, matrix operand order checked by reading, not run on hardware
//
#include <hip/hip_runtime.h>
#include <hip/hip_bf16.h>

typedef _Float16 f16;
typedef __attribute__((ext_vector_type(4)))  _Float16 v4h;
typedef __attribute__((ext_vector_type(8)))  _Float16 v8h;
typedef __attribute__((ext_vector_type(16))) _Float16 v16h;
typedef __attribute__((ext_vector_type(8)))  float    v8f;
typedef __attribute__((ext_vector_type(4)))  unsigned int u32x4;
typedef __attribute__((ext_vector_type(4)))  int i32x4;
typedef __attribute__((ext_vector_type(8)))  int i32x8;

#define BB   2
#define SS   2048
#define DD   1024
#define HH   16
#define HDD  64
#define NTOK (BB*SS)          /* 4096 tokens */

union FragU { v16h v; v8h h[2]; };

// Load a 16x32 f16 operand fragment (A layout; mirrored pattern serves B when
// the operand's output-column axis is contiguous along K in memory).
// base -> element [row0][k0], leading dimension ld (elements).
// Lane L holds row (L&15); k-chunks {kb..kb+7} and {kb+16..kb+23}, kb=8*(L>=16).
__device__ __forceinline__ v16h load_frag(const f16* base, int ld) {
  const int lane = threadIdx.x & 31;
  const int r    = lane & 15;
  const int kb   = (lane >> 4) << 3;
  const f16* p = base + r * ld + kb;
  FragU u;
  u.h[0] = *(const v8h*)(p);
  u.h[1] = *(const v8h*)(p + 16);
  return u.v;
}

__device__ __forceinline__ v8f wmma16(v16h a, v16h b, v8f c) {
  return __builtin_amdgcn_wmma_f32_16x16x32_f16(false, a, false, b, (short)0, c,
                                                false, false);
}

// ---------------------------------------------------------------------------
// Tensor Data Mover: async 2D tile load global -> LDS (f16 elements).
//   lds_off : byte offset of destination tile in LDS
//   gptr    : global address of tile start
//   tile_w  : tile width  (elements, dim0)
//   tile_h  : tile height (rows, dim1)
//   stride  : row stride of the global tensor (elements)
//   tdim1   : #rows of the global tensor (for OOB; we stay in-bounds)
// D# layout per CDNA5 ISA ch.8 (group0 128b, group1 256b; groups 2/3 unused).
// ---------------------------------------------------------------------------
__device__ __forceinline__ void tdm_load_2d(unsigned int lds_off, const f16* gptr,
                                            unsigned int tile_w, unsigned int tile_h,
                                            unsigned int stride, unsigned int tdim1)
{
  const unsigned long long ga = (unsigned long long)(uintptr_t)gptr;
  u32x4 g0;
  g0[0] = 1u;                                              // count=1 (valid), user mode
  g0[1] = lds_off;                                         // lds_addr [63:32]
  g0[2] = (unsigned int)ga;                                // global_addr[31:0]
  g0[3] = ((unsigned int)(ga >> 32) & 0x01ffffffu)         // global_addr[56:32]
        | (2u << 30);                                      // type=2 ("image")
  const unsigned int td0 = stride;                         // tensor_dim0 = stride (no OOB)
  i32x8 g1;
  g1[0] = (int)(1u << 16);                                 // wg_mask=0, data_size=1 (2B)
  g1[1] = (int)(td0 << 16);                                // tensor_dim0[15:0]
  g1[2] = (int)((td0 >> 16) | (tdim1 << 16));              // tensor_dim0[31:16], dim1[15:0]
  g1[3] = (int)((tdim1 >> 16) | (tile_w << 16));           // dim1[31:16], tile_dim0
  g1[4] = (int)tile_h;                                     // tile_dim1 (tile_dim2=0)
  g1[5] = (int)stride;                                     // tensor_dim0_stride[31:0]
  g1[6] = 0;                                               // stride[47:32], dim1_stride lo
  g1[7] = 0;
  i32x4 z4; z4[0] = 0; z4[1] = 0; z4[2] = 0; z4[3] = 0;
#if __clang_major__ >= 23
  i32x8 z8; z8[0]=0; z8[1]=0; z8[2]=0; z8[3]=0; z8[4]=0; z8[5]=0; z8[6]=0; z8[7]=0;
  __builtin_amdgcn_tensor_load_to_lds(g0, g1, z4, z4, z8, 0);
#else
  __builtin_amdgcn_tensor_load_to_lds(g0, g1, z4, z4, 0);
#endif
}

// ---------------------------------------------------------------------------
// f32 -> f16 conversion (n divisible by 4)
// ---------------------------------------------------------------------------
__global__ void cvt_f32_f16(const float* __restrict__ x, f16* __restrict__ y, int n) {
  int i = (blockIdx.x * 256 + threadIdx.x) * 4;
  if (i < n) {
    float4 v = *(const float4*)(x + i);
    v4h o;
    o.x = (f16)v.x; o.y = (f16)v.y; o.z = (f16)v.z; o.w = (f16)v.w;
    *(v4h*)(y + i) = o;
  }
}

// ---------------------------------------------------------------------------
// Projection GEMM: Y[m][n] = sum_k X[m][k] * W[n][k] + bias[n]
//   mode 0/1: + RoPE, store f16 [B,H,S,HD]  (q / k)
//   mode 2:   store f16 transposed [B,H,HD,S]  (v)
//   mode 3:   store f32 [tok][D] to d_out      (output proj)
// Block: 256 threads = 8 waves; wave tile 16(M)x64(N); block tile 128x64.
// Software-pipelined: fragments for K-step k+1 load while step k WMMAs issue.
// ---------------------------------------------------------------------------
__global__ __launch_bounds__(256)
void proj_gemm(const f16* __restrict__ X, const f16* __restrict__ W,
               const float* __restrict__ bias,
               const float* __restrict__ fcos, const float* __restrict__ fsin,
               f16* __restrict__ out16, float* __restrict__ out32, int mode)
{
  const int lane = threadIdx.x & 31;
  const int wv   = threadIdx.x >> 5;
  const int m0   = blockIdx.x * 128 + wv * 16;
  const int n0   = blockIdx.y * 64;
  const int nlo  = lane & 15;
  const int half = lane >> 4;

  const f16* Xb = X + (size_t)m0 * DD;
  const f16* W0 = W + (size_t)(n0 +  0) * DD;
  const f16* W1 = W + (size_t)(n0 + 16) * DD;
  const f16* W2 = W + (size_t)(n0 + 32) * DD;
  const f16* W3 = W + (size_t)(n0 + 48) * DD;

  v8f acc[4] = {};
  v16h a  = load_frag(Xb, DD);
  v16h b0 = load_frag(W0, DD);
  v16h b1 = load_frag(W1, DD);
  v16h b2 = load_frag(W2, DD);
  v16h b3 = load_frag(W3, DD);

  for (int k0 = 0; k0 < DD; k0 += 32) {
    const int kn = (k0 + 32 < DD) ? (k0 + 32) : 0;   // harmless wrap on last step
    v16h an = load_frag(Xb + kn, DD);
    v16h c0 = load_frag(W0 + kn, DD);
    v16h c1 = load_frag(W1 + kn, DD);
    v16h c2 = load_frag(W2 + kn, DD);
    v16h c3 = load_frag(W3 + kn, DD);
    acc[0] = wmma16(a, b0, acc[0]);
    acc[1] = wmma16(a, b1, acc[1]);
    acc[2] = wmma16(a, b2, acc[2]);
    acc[3] = wmma16(a, b3, acc[3]);
    a = an; b0 = c0; b1 = c1; b2 = c2; b3 = c3;
  }

#pragma unroll
  for (int c = 0; c < 4; ++c) {
    const int n  = n0 + c * 16 + nlo;
    const float bn = bias[n];
    const int h  = n >> 6;
    const int hd = n & 63;
#pragma unroll
    for (int r = 0; r < 8; ++r) {
      const int m  = m0 + r + half * 8;          // global token index
      const int bb = m >> 11;                    // m / S
      const int sp = m & 2047;                   // m % S
      float v = acc[c][r] + bn;
      if (mode <= 1) {
        const int pair = hd >> 1;
        const float co = fcos[sp * (HDD / 2) + pair];
        const float si = fsin[sp * (HDD / 2) + pair];
        const float partner = __shfl_xor(v, 1);  // paired feature lives in lane^1
        v = (hd & 1) ? (partner * si + v * co)   // odd: x_r*sin + x_i*cos
                     : (v * co - partner * si);  // even: x_r*cos - x_i*sin
      }
      if (mode == 3) {
        out32[m * DD + n] = v;
      } else if (mode == 2) {
        out16[((bb * HH + h) * HDD + hd) * SS + sp] = (f16)v;   // [B,H,HD,S]
      } else {
        out16[((bb * HH + h) * SS + sp) * HDD + hd] = (f16)v;   // [B,H,S,HD]
      }
    }
  }
}

// ---------------------------------------------------------------------------
// Causal flash attention with TDM-staged K/V tiles.
// grid = (S/64, B*H), block = 128 (4 waves). Wave w owns rows q0..q0+15.
// K/V 64x64 f16 tiles are DMA'd into double-buffered LDS by the Tensor Data
// Mover (wave 0 issues, TENSORcnt + barrier synchronize), WMMAs read LDS.
// qr,kr: [B,H,S,HD] f16.  vt: [B,H,HD,S] f16.  O: [B,S,D] f16.
// ---------------------------------------------------------------------------
__global__ __launch_bounds__(128)
void flash_attn(const f16* __restrict__ qr, const f16* __restrict__ kr,
                const f16* __restrict__ vt, f16* __restrict__ O)
{
  __shared__ f16 Kt[2][64][64];      // [buf][key][hd]
  __shared__ f16 Vs[2][64][64];      // [buf][hd][key]
  __shared__ f16 Pl[4][16][64];      // per-wave P staging (C-frag -> A-frag)

  const int lane = threadIdx.x & 31;
  const int wv   = threadIdx.x >> 5;
  const int bh   = blockIdx.y;
  const int b    = bh >> 4;
  const int h    = bh & 15;
  const int q0   = blockIdx.x * 64 + wv * 16;
  const int nlo  = lane & 15;
  const int half = lane >> 4;

  const f16* qbase = qr + (size_t)(bh * SS + q0) * HDD;
  const f16* kbase = kr + (size_t)bh * SS * HDD;
  const f16* vbase = vt + (size_t)bh * HDD * SS;

  const unsigned int ktOff = (unsigned int)(uintptr_t)&Kt[0][0][0];
  const unsigned int vsOff = (unsigned int)(uintptr_t)&Vs[0][0][0];

  const v16h qa0 = load_frag(qbase + 0,  HDD);
  const v16h qa1 = load_frag(qbase + 32, HDD);

  v8f accv[4] = {};
  float mrow[8], lrow[8];
#pragma unroll
  for (int r = 0; r < 8; ++r) { mrow[r] = -1e30f; lrow[r] = 0.0f; }

  const int jmax = blockIdx.x;       // uniform trip count: barriers are legal
  for (int j = 0; j <= jmax; ++j) {
    const int buf = j & 1;
    if (wv == 0) {
      if (j == 0) {
        tdm_load_2d(ktOff, kbase, 64, 64, HDD, SS);
        tdm_load_2d(vsOff, vbase, 64, 64, SS, HDD);
      }
      if (j < jmax) {                 // prefetch next tiles into other buffer
        const unsigned int nb = (unsigned int)((j + 1) & 1) * (64u * 64u * 2u);
        tdm_load_2d(ktOff + nb, kbase + (size_t)(j + 1) * 64 * HDD, 64, 64, HDD, SS);
        tdm_load_2d(vsOff + nb, vbase + (j + 1) * 64,               64, 64, SS, HDD);
        __builtin_amdgcn_s_wait_tensorcnt(2);   // tile j complete (in-order TDM)
      } else {
        __builtin_amdgcn_s_wait_tensorcnt(0);
      }
    }
    __syncthreads();                  // publish tile j to all waves
    const f16* ktile = &Kt[buf][0][0];
    const f16* vtile = &Vs[buf][0][0];

    // ---- S = (Q K^T) * scale, diagonal-tile causal mask ----
    v8f s[4];
    {
      v16h k00 = load_frag(ktile + 0 * 16 * 64 + 0, 64);
      v16h k01 = load_frag(ktile + 0 * 16 * 64 + 32, 64);
      v16h k10 = load_frag(ktile + 1 * 16 * 64 + 0, 64);
      v16h k11 = load_frag(ktile + 1 * 16 * 64 + 32, 64);
      v16h k20 = load_frag(ktile + 2 * 16 * 64 + 0, 64);
      v16h k21 = load_frag(ktile + 2 * 16 * 64 + 32, 64);
      v16h k30 = load_frag(ktile + 3 * 16 * 64 + 0, 64);
      v16h k31 = load_frag(ktile + 3 * 16 * 64 + 32, 64);
      v8f z0 = {}, z1 = {}, z2 = {}, z3 = {};
      z0 = wmma16(qa0, k00, z0); z0 = wmma16(qa1, k01, z0);
      z1 = wmma16(qa0, k10, z1); z1 = wmma16(qa1, k11, z1);
      z2 = wmma16(qa0, k20, z2); z2 = wmma16(qa1, k21, z2);
      z3 = wmma16(qa0, k30, z3); z3 = wmma16(qa1, k31, z3);
      s[0] = z0; s[1] = z1; s[2] = z2; s[3] = z3;
    }
    const bool diag = (j == jmax);
#pragma unroll
    for (int c = 0; c < 4; ++c)
#pragma unroll
      for (int r = 0; r < 8; ++r) {
        float sv = s[c][r] * 0.125f;
        if (diag) {
          const int ng = j * 64 + c * 16 + nlo;
          const int mg = q0 + r + half * 8;
          if (ng > mg) sv = -1e10f;
        }
        s[c][r] = sv;
      }

    // ---- online softmax: row reductions inside each 16-lane half ----
    float mnew[8], alpha[8];
#pragma unroll
    for (int r = 0; r < 8; ++r) {
      float mx = fmaxf(fmaxf(s[0][r], s[1][r]), fmaxf(s[2][r], s[3][r]));
      mx = fmaxf(mx, __shfl_xor(mx, 1));
      mx = fmaxf(mx, __shfl_xor(mx, 2));
      mx = fmaxf(mx, __shfl_xor(mx, 4));
      mx = fmaxf(mx, __shfl_xor(mx, 8));
      mnew[r]  = fmaxf(mrow[r], mx);
      alpha[r] = __expf(mrow[r] - mnew[r]);
      mrow[r]  = mnew[r];
    }
#pragma unroll
    for (int c = 0; c < 4; ++c)
#pragma unroll
      for (int r = 0; r < 8; ++r)
        s[c][r] = __expf(s[c][r] - mnew[r]);
#pragma unroll
    for (int r = 0; r < 8; ++r) {
      float t = s[0][r] + s[1][r] + s[2][r] + s[3][r];
      t += __shfl_xor(t, 1);
      t += __shfl_xor(t, 2);
      t += __shfl_xor(t, 4);
      t += __shfl_xor(t, 8);
      lrow[r] = lrow[r] * alpha[r] + t;
    }
#pragma unroll
    for (int c = 0; c < 4; ++c)
#pragma unroll
      for (int r = 0; r < 8; ++r)
        accv[c][r] *= alpha[r];

    // ---- re-layout P via per-wave LDS tile (C-frag -> A-frag) ----
#pragma unroll
    for (int c = 0; c < 4; ++c)
#pragma unroll
      for (int r = 0; r < 8; ++r)
        Pl[wv][r + half * 8][c * 16 + nlo] = (f16)s[c][r];
    __syncthreads();

    // ---- O += P @ V ----
    {
      const f16* pb = &Pl[wv][0][0];
      const v16h pa0 = load_frag(pb + 0,  64);
      const v16h pa1 = load_frag(pb + 32, 64);
#pragma unroll
      for (int c = 0; c < 4; ++c) {
        accv[c] = wmma16(pa0, load_frag(vtile + (c * 16) * 64 + 0,  64), accv[c]);
        accv[c] = wmma16(pa1, load_frag(vtile + (c * 16) * 64 + 32, 64), accv[c]);
      }
    }
    __syncthreads();                 // seal tile j before its buffer is re-DMAd
  }

  // ---- normalize and write O as [B,S,D] f16 ----
#pragma unroll
  for (int c = 0; c < 4; ++c)
#pragma unroll
    for (int r = 0; r < 8; ++r) {
      const int m = q0 + r + half * 8;
      const float o = accv[c][r] / lrow[r];
      O[((size_t)b * SS + m) * DD + h * 64 + c * 16 + nlo] = (f16)o;
    }
}

// ---------------------------------------------------------------------------
// Host-side orchestration
// ---------------------------------------------------------------------------
extern "C" void kernel_launch(void* const* d_in, const int* in_sizes, int n_in,
                              void* d_out, int out_size, void* d_ws, size_t ws_size,
                              hipStream_t stream) {
  // inputs: 0 Q, 1 K, 2 V, 3 mask, 4 freqs_cos, 5 freqs_sin,
  //         6 wq, 7 bq, 8 wk, 9 bk, 10 wv, 11 bv, 12 wo, 13 bo
  const float* Q    = (const float*)d_in[0];
  const float* K    = (const float*)d_in[1];
  const float* V    = (const float*)d_in[2];
  const float* fcos = (const float*)d_in[4];
  const float* fsin = (const float*)d_in[5];
  const float* wq   = (const float*)d_in[6];
  const float* bq   = (const float*)d_in[7];
  const float* wk   = (const float*)d_in[8];
  const float* bk   = (const float*)d_in[9];
  const float* wvp  = (const float*)d_in[10];
  const float* bv   = (const float*)d_in[11];
  const float* wo   = (const float*)d_in[12];
  const float* bo   = (const float*)d_in[13];
  float* out = (float*)d_out;

  const size_t ACT = (size_t)NTOK * DD;          // 4M elems
  const size_t WN  = (size_t)DD * DD;            // 1M elems
  const size_t HN  = (size_t)BB * HH * SS * HDD; // 4M elems

  f16* p    = (f16*)d_ws;
  f16* hQ   = p; p += ACT;
  f16* hK   = p; p += ACT;
  f16* hV   = p; p += ACT;
  f16* hWq  = p; p += WN;
  f16* hWk  = p; p += WN;
  f16* hWv  = p; p += WN;
  f16* hWo  = p; p += WN;
  f16* qro  = p; p += HN;   // [B,H,S,HD]
  f16* kro  = p; p += HN;   // [B,H,S,HD]
  f16* vtr  = p; p += HN;   // [B,H,HD,S]
  f16* Oat  = p; p += ACT;  // [B,S,D]

  // 1) convert activations + weights to f16
  {
    dim3 blk(256);
    cvt_f32_f16<<<dim3((ACT / 4 + 255) / 256), blk, 0, stream>>>(Q,   hQ,  (int)ACT);
    cvt_f32_f16<<<dim3((ACT / 4 + 255) / 256), blk, 0, stream>>>(K,   hK,  (int)ACT);
    cvt_f32_f16<<<dim3((ACT / 4 + 255) / 256), blk, 0, stream>>>(V,   hV,  (int)ACT);
    cvt_f32_f16<<<dim3((WN  / 4 + 255) / 256), blk, 0, stream>>>(wq,  hWq, (int)WN);
    cvt_f32_f16<<<dim3((WN  / 4 + 255) / 256), blk, 0, stream>>>(wk,  hWk, (int)WN);
    cvt_f32_f16<<<dim3((WN  / 4 + 255) / 256), blk, 0, stream>>>(wvp, hWv, (int)WN);
    cvt_f32_f16<<<dim3((WN  / 4 + 255) / 256), blk, 0, stream>>>(wo,  hWo, (int)WN);
  }

  // 2) q/k/v projections (+bias, +RoPE on q/k), head-major layouts
  {
    dim3 grid(NTOK / 128, DD / 64), blk(256);
    proj_gemm<<<grid, blk, 0, stream>>>(hQ, hWq, bq, fcos, fsin, qro, nullptr, 0);
    proj_gemm<<<grid, blk, 0, stream>>>(hK, hWk, bk, fcos, fsin, kro, nullptr, 1);
    proj_gemm<<<grid, blk, 0, stream>>>(hV, hWv, bv, fcos, fsin, vtr, nullptr, 2);
  }

  // 3) causal flash attention (TDM-staged K/V tiles)
  {
    dim3 grid(SS / 64, BB * HH), blk(128);
    flash_attn<<<grid, blk, 0, stream>>>(qro, kro, vtr, Oat);
  }

  // 4) output projection -> f32 d_out
  {
    dim3 grid(NTOK / 128, DD / 64), blk(256);
    proj_gemm<<<grid, blk, 0, stream>>>(Oat, hWo, bo, nullptr, nullptr, nullptr, out, 3);
  }
}
